// NucleusXMultiScaleRetention_64939905516227
// MI455X (gfx1250) — matrix operand order repaired and checked by reading, not compiled, not run on hardware
//
#include <hip/hip_runtime.h>

#define B_ 2
#define T_ 2048
#define E_ 2048
#define V_ 4096
#define NH_ 8
#define KD_ 256
#define HD_ 512
#define MT_ (B_ * T_)  // 4096 rows total

typedef __bf16 bf16;
typedef __attribute__((ext_vector_type(16))) __bf16 v16bf;
typedef __attribute__((ext_vector_type(8)))  __bf16 v8bf;
typedef __attribute__((ext_vector_type(8)))  float  v8f;

// ---------------------------------------------------------------------------
// WMMA fragment helpers (wave32; A/B use this code's contiguous k order, not the ISA table order)
// A (16x32 bf16): lane L holds row m=L%16; halves 0..15 = K (L/16)*16+0..15.
// B (32x16 bf16): lane L holds col n=L%16; halves 0..15 = K (L/16)*16+0..15.
// C/D (16x16 f32): VGPR r: row = r + 8*(L/16), col = L%16.
// ---------------------------------------------------------------------------
static __device__ __forceinline__ v16bf load_a_frag(const bf16* base, int ld, int lane) {
  const bf16* p = base + (size_t)(lane & 15) * (size_t)ld + ((lane >> 4) << 4);
  v8bf lo = *(const v8bf*)p;
  v8bf hi = *(const v8bf*)(p + 8);
  return __builtin_shufflevector(lo, hi, 0,1,2,3,4,5,6,7,8,9,10,11,12,13,14,15);
}
static __device__ __forceinline__ v16bf load_b_frag(const bf16* base, int ld, int lane) {
  const bf16* p = base + (size_t)(lane & 15) * (size_t)ld + ((lane >> 4) << 4);
  return *(const v16bf*)p;
}
static __device__ __forceinline__ v8f wmma_bf16(v16bf a, v16bf b, v8f c) {
  return __builtin_amdgcn_wmma_f32_16x16x32_bf16(false, a, false, b, (short)0, c,
                                                 false, false);
}
// Generic shared pointer -> 32-bit LDS offset (LDS aperture: addr[31:0] = offset)
static __device__ __forceinline__ unsigned lds_addr32(const void* p) {
  return (unsigned)(size_t)p;
}
// Async global -> LDS copy, 16B per lane, tracked by ASYNCcnt.
static __device__ __forceinline__ void async_copy_b128(unsigned lds, const void* gp) {
  asm volatile("global_load_async_to_lds_b128 %0, %1, off"
               :: "v"(lds), "v"(gp) : "memory");
}

// ---------------------------------------------------------------------------
// f32 -> bf16 convert
// ---------------------------------------------------------------------------
__global__ void k_f32_to_bf16(const float* __restrict__ src, bf16* __restrict__ dst,
                              size_t n) {
  size_t i = (size_t)blockIdx.x * blockDim.x + threadIdx.x;
  if (i < n) dst[i] = (bf16)src[i];
}

// ---------------------------------------------------------------------------
// bf16 WMMA GEMM with async double-buffered LDS staging:
//   C[M,N] = A[M,K] * W[N,K]^T   (f32 accumulate/output)
// Block = 256 threads (8 waves, 4x2), workgroup tile 128x64, wave tile 32x32,
// K-step 64.  Tiles staged via GLOBAL_LOAD_ASYNC_TO_LDS_B128 (ASYNCcnt),
// consumed via ds_load fragments.  Rows padded +8 bf16 to spread LDS banks.
// ---------------------------------------------------------------------------
#define KSTEP 64
#define ALD   (KSTEP + 8)  // 72 bf16 = 144B row pitch

__global__ __launch_bounds__(256) void k_gemm_bf16(const bf16* __restrict__ A,
                                                   const bf16* __restrict__ W,
                                                   float* __restrict__ C,
                                                   int M, int N, int K) {
  __shared__ bf16 sA[2][128 * ALD];  // 2 x 18KB
  __shared__ bf16 sW[2][64 * ALD];   // 2 x  9KB

  const int tid = threadIdx.x;
  const int lane = tid & 31;
  const int w = tid >> 5;
  const int wm = w & 3, wn = w >> 2;
  const int mBlk = blockIdx.y * 128;
  const int nBlk = blockIdx.x * 64;

  const int lrow = tid >> 3;          // 0..31
  const int lchunk = (tid & 7) * 8;   // bf16 offset of this thread's 16B chunk

  auto issue = [&](int p, int k0) {
    // A tile 128 x 64: 4 async b128 issues per thread
#pragma unroll
    for (int i = 0; i < 4; ++i) {
      const int r = lrow + 32 * i;
      const bf16* g = A + (size_t)(mBlk + r) * K + k0 + lchunk;
      async_copy_b128(lds_addr32(&sA[p][r * ALD + lchunk]), g);
    }
    // W tile 64 x 64: 2 issues per thread
#pragma unroll
    for (int i = 0; i < 2; ++i) {
      const int r = lrow + 32 * i;
      const bf16* g = W + (size_t)(nBlk + r) * K + k0 + lchunk;
      async_copy_b128(lds_addr32(&sW[p][r * ALD + lchunk]), g);
    }
  };

  v8f acc00 = {}, acc01 = {}, acc10 = {}, acc11 = {};
  const int nIter = K / KSTEP;

  issue(0, 0);
  for (int it = 0; it < nIter; ++it) {
    const int p = it & 1;
    if (it + 1 < nIter) {
      issue(p ^ 1, (it + 1) * KSTEP);
      asm volatile("s_wait_asynccnt 0x6" ::: "memory");  // current buffer done
    } else {
      asm volatile("s_wait_asynccnt 0x0" ::: "memory");
    }
    __syncthreads();

    const bf16* tA = &sA[p][wm * 32 * ALD];
    const bf16* tW = &sW[p][wn * 32 * ALD];
#pragma unroll
    for (int kk = 0; kk < KSTEP; kk += 32) {
      v16bf a0 = load_a_frag(tA + kk, ALD, lane);
      v16bf a1 = load_a_frag(tA + 16 * ALD + kk, ALD, lane);
      v16bf b0 = load_b_frag(tW + kk, ALD, lane);
      v16bf b1 = load_b_frag(tW + 16 * ALD + kk, ALD, lane);
      acc00 = wmma_bf16(a0, b0, acc00);
      acc01 = wmma_bf16(a0, b1, acc01);
      acc10 = wmma_bf16(a1, b0, acc10);
      acc11 = wmma_bf16(a1, b1, acc11);
    }
    __syncthreads();  // safe to overwrite buffer p next round
  }

  const int colb = nBlk + wn * 32 + (lane & 15);
  const int rowb = mBlk + wm * 32 + ((lane >> 4) << 3);
#pragma unroll
  for (int r = 0; r < 8; ++r) {
    C[(size_t)(rowb + r) * N + colb]           = acc00[r];
    C[(size_t)(rowb + r) * N + colb + 16]      = acc01[r];
    C[(size_t)(rowb + 16 + r) * N + colb]      = acc10[r];
    C[(size_t)(rowb + 16 + r) * N + colb + 16] = acc11[r];
  }
}

// ---------------------------------------------------------------------------
// RoPE (theta shift) on q/k (+ fold in k-scaling), relayout to [b,h,t,kd] bf16,
// and V relayout to transposed [b,h,hd,t] bf16.
// grid = (B*T, NH), block = 128 (one thread per even/odd pair of KD)
// ---------------------------------------------------------------------------
__global__ __launch_bounds__(128) void k_rope_relayout(
    const float* __restrict__ qf, const float* __restrict__ kf,
    const float* __restrict__ vf, const float* __restrict__ sinp,
    const float* __restrict__ cosp, bf16* __restrict__ qr,
    bf16* __restrict__ kr, bf16* __restrict__ vbT) {
  const int bt = blockIdx.x;          // b*T + t
  const int h  = blockIdx.y;
  const int t  = bt & (T_ - 1);
  const int b  = bt >> 11;            // T_ = 2048
  const int i  = threadIdx.x;         // pair index 0..127
  const int d0 = 2 * i, d1 = 2 * i + 1;

  const size_t src = (size_t)bt * E_ + (size_t)h * KD_ + d0;
  const float x0q = qf[src], x1q = qf[src + 1];
  const float x0k = kf[src], x1k = kf[src + 1];
  const float s0 = sinp[(size_t)t * KD_ + d0], s1 = sinp[(size_t)t * KD_ + d1];
  const float c0 = cosp[(size_t)t * KD_ + d0], c1 = cosp[(size_t)t * KD_ + d1];

  const float SC = 0.0625f;  // KD^-0.5 = 1/16
  const float q0 = x0q * c0 - x1q * s0;
  const float q1 = x1q * c1 + x0q * s1;
  const float k0 = (x0k * c0 - x1k * s0) * SC;
  const float k1 = (x1k * c1 + x0k * s1) * SC;

  const size_t dst = (((size_t)(b * NH_ + h)) * T_ + t) * KD_ + d0;
  qr[dst] = (bf16)q0; qr[dst + 1] = (bf16)q1;
  kr[dst] = (bf16)k0; kr[dst + 1] = (bf16)k1;

#pragma unroll
  for (int j = 0; j < 4; ++j) {
    const int d = i + 128 * j;  // 0..511
    const float v = vf[(size_t)bt * V_ + (size_t)h * HD_ + d];
    vbT[(((size_t)(b * NH_ + h)) * HD_ + d) * T_ + t] = (bf16)v;
  }
}

// ---------------------------------------------------------------------------
// Streaming retention: per (b,h), 64-row block of output rows.
// 8 waves = 4 (M subtiles of 16 rows) x 2 (HD halves of 256 cols).
// Q panel (16x256 per wave) hoisted into registers once (8 A-fragments).
// Loop over s in tiles of 32:
//   wn==0 waves: S = Qr x Kr^T via WMMA, apply decay mask, accumulate row
//                sums, stage bf16 S tile in LDS (doubles as layout transpose).
//   all waves:   O(16x256) += S(16x32) x V(32x256) via WMMA (V pre-transposed).
// Epilogue: scale rows by 1/max(|rowsum|,1), write f32 obuf[b,t,h,hd].
// ---------------------------------------------------------------------------
__global__ __launch_bounds__(256) void k_retention(
    const bf16* __restrict__ qr, const bf16* __restrict__ kr,
    const bf16* __restrict__ vbT, const float* __restrict__ dmask,
    float* __restrict__ obuf) {
  __shared__ bf16 sS[4][16 * 32];
  __shared__ float sDen[64];

  const int lane = threadIdx.x & 31;
  const int w = threadIdx.x >> 5;
  const int wm = w & 3, wn = w >> 2;
  const int bh = blockIdx.y;
  const int b = bh >> 3, h = bh & 7;
  const int m0 = blockIdx.x * 64;
  const int mRow = m0 + wm * 16;

  const bf16* Q  = qr  + (size_t)bh * T_ * KD_;
  const bf16* Kr = kr  + (size_t)bh * T_ * KD_;
  const bf16* Vt = vbT + (size_t)bh * HD_ * T_;
  const float* DM = dmask + (size_t)h * T_ * T_;

  v8f o[16];
#pragma unroll
  for (int i = 0; i < 16; ++i) o[i] = (v8f){};
  v8f rsum = {};

  const int mloc8 = (lane >> 4) << 3;  // 0 or 8
  const int nloc = lane & 15;

  // Hoist the Q panel for this wave's 16 rows: 8 A-fragments (loop invariant).
  v16bf qa[8];
  if (wn == 0) {
    const bf16* Qb = Q + (size_t)mRow * KD_;
#pragma unroll
    for (int kk = 0; kk < 8; ++kk) qa[kk] = load_a_frag(Qb + kk * 32, KD_, lane);
  }

  for (int s0 = 0; s0 < T_; s0 += 32) {
    if (wn == 0) {
      v8f sa0 = {}, sa1 = {};
      const bf16* Kb = Kr + (size_t)s0 * KD_;
#pragma unroll
      for (int kk = 0; kk < 8; ++kk) {
        v16bf b0 = load_b_frag(Kb + kk * 32, KD_, lane);
        v16bf b1 = load_b_frag(Kb + (size_t)16 * KD_ + kk * 32, KD_, lane);
        sa0 = wmma_bf16(qa[kk], b0, sa0);
        sa1 = wmma_bf16(qa[kk], b1, sa1);
      }
      // decay mask, row-sum accumulation, stage bf16 S tile to LDS
#pragma unroll
      for (int r = 0; r < 8; ++r) {
        const int ml = r + mloc8;
        const size_t dmrow = (size_t)(mRow + ml) * T_ + s0;
        float v0 = sa0[r] * DM[dmrow + nloc];
        float v1 = sa1[r] * DM[dmrow + 16 + nloc];
        rsum[r] += v0 + v1;
        sS[wm][ml * 32 + nloc]      = (bf16)v0;
        sS[wm][ml * 32 + 16 + nloc] = (bf16)v1;
      }
    }
    __syncthreads();

    v16bf a = load_a_frag(&sS[wm][0], 32, lane);
#pragma unroll
    for (int nt = 0; nt < 16; ++nt) {
      const int hd = wn * 256 + nt * 16;
      v16bf bv = load_b_frag(Vt + (size_t)hd * T_ + s0, T_, lane);
      o[nt] = wmma_bf16(a, bv, o[nt]);
    }
    __syncthreads();
  }

  // reduce row sums across the 16-lane halves, publish denominators
  if (wn == 0) {
#pragma unroll
    for (int r = 0; r < 8; ++r) {
      float v = rsum[r];
      v += __shfl_xor(v, 1);
      v += __shfl_xor(v, 2);
      v += __shfl_xor(v, 4);
      v += __shfl_xor(v, 8);
      if ((lane & 15) == 0) sDen[wm * 16 + r + mloc8] = v;
    }
  }
  __syncthreads();

#pragma unroll
  for (int r = 0; r < 8; ++r) {
    const int ml = r + mloc8;
    const float scale = 1.0f / fmaxf(fabsf(sDen[wm * 16 + ml]), 1.0f);
    const int trow = mRow + ml;
#pragma unroll
    for (int nt = 0; nt < 16; ++nt) {
      const int hd = wn * 256 + nt * 16 + nloc;
      obuf[(((size_t)b * T_ + trow) * NH_ + h) * HD_ + hd] = o[nt][r] * scale;
    }
  }
}

// ---------------------------------------------------------------------------
// Per-(b,t,head) RMS norm over HD + SiLU gate; write bf16 gated[b,t,V].
// grid = (B*T), block = 256 (wave per head).
// ---------------------------------------------------------------------------
__global__ __launch_bounds__(256) void k_norm_gate(const float* __restrict__ obuf,
                                                   const float* __restrict__ gf,
                                                   bf16* __restrict__ gated) {
  const int bt = blockIdx.x;
  const int lane = threadIdx.x & 31;
  const int h = threadIdx.x >> 5;
  const float* op = obuf + ((size_t)bt * NH_ + h) * HD_;

  float vals[16];
  float ss = 0.0f;
#pragma unroll
  for (int j = 0; j < 16; ++j) {
    const float v = op[j * 32 + lane];
    vals[j] = v;
    ss += v * v;
  }
  ss += __shfl_xor(ss, 1);
  ss += __shfl_xor(ss, 2);
  ss += __shfl_xor(ss, 4);
  ss += __shfl_xor(ss, 8);
  ss += __shfl_xor(ss, 16);
  const float rinv = rsqrtf(ss * (1.0f / (float)HD_) + 1e-6f);

  const float* gp = gf + (size_t)bt * V_ + (size_t)h * HD_;
  bf16* out = gated + (size_t)bt * V_ + (size_t)h * HD_;
#pragma unroll
  for (int j = 0; j < 16; ++j) {
    const int d = j * 32 + lane;
    const float g = gp[d];
    const float sig = 1.0f / (1.0f + __expf(-g));
    out[d] = (bf16)(g * sig * vals[j] * rinv);
  }
}

// ---------------------------------------------------------------------------
// Host launch
// ---------------------------------------------------------------------------
extern "C" void kernel_launch(void* const* d_in, const int* in_sizes, int n_in,
                              void* d_out, int out_size, void* d_ws, size_t ws_size,
                              hipStream_t stream) {
  const float* hs    = (const float*)d_in[0];
  const float* sinp  = (const float*)d_in[1];
  const float* cosp  = (const float*)d_in[2];
  const float* dmask = (const float*)d_in[3];
  const float* wq    = (const float*)d_in[4];
  const float* wk    = (const float*)d_in[5];
  const float* wv    = (const float*)d_in[6];
  const float* wg    = (const float*)d_in[7];
  const float* wout  = (const float*)d_in[8];
  float* outp = (float*)d_out;

  char* ws = (char*)d_ws;
  size_t off = 0;
  auto alloc = [&](size_t bytes) -> void* {
    void* p = ws + off;
    off += (bytes + 255) & ~(size_t)255;
    return p;
  };

  bf16* hsb   = (bf16*)alloc((size_t)MT_ * E_ * 2);
  bf16* wqb   = (bf16*)alloc((size_t)E_ * E_ * 2);
  bf16* wkb   = (bf16*)alloc((size_t)E_ * E_ * 2);
  bf16* wvb   = (bf16*)alloc((size_t)V_ * E_ * 2);
  bf16* wgb   = (bf16*)alloc((size_t)V_ * E_ * 2);
  bf16* woutb = (bf16*)alloc((size_t)E_ * V_ * 2);
  float* qf   = (float*)alloc((size_t)MT_ * E_ * 4);
  float* kf   = (float*)alloc((size_t)MT_ * E_ * 4);
  float* vf   = (float*)alloc((size_t)MT_ * V_ * 4);
  float* gf   = (float*)alloc((size_t)MT_ * V_ * 4);
  bf16* qrb   = (bf16*)alloc((size_t)MT_ * E_ * 2);
  bf16* krb   = (bf16*)alloc((size_t)MT_ * E_ * 2);
  bf16* vbT   = (bf16*)alloc((size_t)MT_ * V_ * 2);
  float* obuf = (float*)alloc((size_t)MT_ * V_ * 4);
  bf16* gated = (bf16*)alloc((size_t)MT_ * V_ * 2);
  (void)ws_size; (void)in_sizes; (void)n_in; (void)out_size;

  auto conv = [&](const float* s, bf16* d, size_t n) {
    k_f32_to_bf16<<<dim3((unsigned)((n + 255) / 256)), 256, 0, stream>>>(s, d, n);
  };
  conv(hs,   hsb,   (size_t)MT_ * E_);
  conv(wq,   wqb,   (size_t)E_ * E_);
  conv(wk,   wkb,   (size_t)E_ * E_);
  conv(wv,   wvb,   (size_t)V_ * E_);
  conv(wg,   wgb,   (size_t)V_ * E_);
  conv(wout, woutb, (size_t)E_ * V_);

  // q/k/v/g projections: C = hs * W^T
  k_gemm_bf16<<<dim3(E_ / 64, MT_ / 128), 256, 0, stream>>>(hsb, wqb, qf, MT_, E_, E_);
  k_gemm_bf16<<<dim3(E_ / 64, MT_ / 128), 256, 0, stream>>>(hsb, wkb, kf, MT_, E_, E_);
  k_gemm_bf16<<<dim3(V_ / 64, MT_ / 128), 256, 0, stream>>>(hsb, wvb, vf, MT_, V_, E_);
  k_gemm_bf16<<<dim3(V_ / 64, MT_ / 128), 256, 0, stream>>>(hsb, wgb, gf, MT_, V_, E_);

  k_rope_relayout<<<dim3(MT_, NH_), 128, 0, stream>>>(qf, kf, vf, sinp, cosp,
                                                      qrb, krb, vbT);

  k_retention<<<dim3(T_ / 64, B_ * NH_), 256, 0, stream>>>(qrb, krb, vbT, dmask, obuf);

  k_norm_gate<<<dim3(MT_), 256, 0, stream>>>(obuf, gf, gated);

  // final projection: out = gated * wout^T  (M=4096, N=2048, K=4096)
  k_gemm_bf16<<<dim3(E_ / 64, MT_ / 128), 256, 0, stream>>>(gated, woutb, outp,
                                                            MT_, E_, V_);
}
